// OnlineFlashMTPModel_58866821759486
// MI455X (gfx1250) — matrix operand order repaired and checked by reading, not compiled, not run on hardware
//
#include <hip/hip_runtime.h>
#include <stdint.h>

typedef __attribute__((ext_vector_type(16))) __bf16 v16bf;
typedef __attribute__((ext_vector_type(8)))  __bf16 v8bf;
typedef __attribute__((ext_vector_type(8)))  float  v8f;
typedef __attribute__((ext_vector_type(4)))  unsigned int v4u;
typedef __attribute__((ext_vector_type(4)))  int v4i;

#define B_DIM   2
#define S_DIM   4096
#define D_DIM   1024
#define V_DIM   32000
#define NA      256
#define BS      16
#define L_TGT   4
#define G_TOK   20      // L_TGT + BS
#define NH      16
#define DH      64
#define MASK_ID 31999

// GEMM block tiling: 8 waves/block, each wave 32(M)x64(N), block = 256(M) x 64(N)
#define BK      32      // k per step
#define WM      32      // rows per wave
#define BM      256     // rows per block
#define BN      64      // cols per block
#define SLAB    (BN*BK) // halves per LDS B slab (4 KB)

__device__ __forceinline__ unsigned short f2bf(float f) {
    unsigned u = __float_as_uint(f);
    u += 0x7FFFu + ((u >> 16) & 1u);          // round-to-nearest-even
    return (unsigned short)(u >> 16);
}
__device__ __forceinline__ float bf2f(unsigned short h) {
    return __uint_as_float(((unsigned)h) << 16);
}

// ---------------- async global->LDS staging (CDNA5), with sync fallback ----------------
#if __has_builtin(__builtin_amdgcn_global_load_async_to_lds_b128)
__device__ __forceinline__ void copy_slab16(const unsigned short* g, unsigned short* l) {
    // one B128 per lane, tracked by ASYNCcnt.  Param 0 is v4i addrspace(1)*,
    // param 1 is the LDS destination (addrspace(3)); generic LDS addr low 32 bits
    // are the LDS byte offset, so truncate through unsigned.
    __builtin_amdgcn_global_load_async_to_lds_b128(
        (__attribute__((address_space(1))) v4i*)(uintptr_t)g,
        (__attribute__((address_space(3))) v4i*)(unsigned)(uintptr_t)l, 0, 0);
}
#define HAVE_ASYNC_LDS 1
#else
__device__ __forceinline__ void copy_slab16(const unsigned short* g, unsigned short* l) {
    *(v4u*)l = *(const v4u*)g;                // global_load_b128 + ds_store_b128
}
#define HAVE_ASYNC_LDS 0
#endif

__device__ __forceinline__ void wait_async_le(int n) {
#if HAVE_ASYNC_LDS
#if __has_builtin(__builtin_amdgcn_s_wait_asynccnt)
    if (n == 0) __builtin_amdgcn_s_wait_asynccnt(0);
    else        __builtin_amdgcn_s_wait_asynccnt(1);
#else
    if (n == 0) asm volatile("s_wait_asynccnt 0x0" ::: "memory");
    else        asm volatile("s_wait_asynccnt 0x1" ::: "memory");
#endif
#else
    (void)n;
#endif
}

// ---------------- elementwise conversions ----------------
__global__ void k_cvt(const float* __restrict__ src, unsigned short* __restrict__ dst, int n) {
    int i = blockIdx.x * blockDim.x + threadIdx.x;
    if (i < n) dst[i] = f2bf(src[i]);
}

// W (K x N) f32 -> Wt (N x K) bf16  (so WMMA B-fragments are contiguous per lane)
__global__ void k_cvt_t(const float* __restrict__ src, unsigned short* __restrict__ dst, int n) {
    int i = blockIdx.x * blockDim.x + threadIdx.x;
    if (i >= n) return;
    int nn = i / D_DIM, kk = i % D_DIM;
    dst[i] = f2bf(src[kk * D_DIM + nn]);
}

// ---------------- build x = [CHS | noise-block embeds] in bf16 ----------------
__global__ void k_build_x(const int* __restrict__ input_ids,
                          const float* __restrict__ hidden,     // (5,B,S,D)
                          const int* __restrict__ anchors,      // (B,NA)
                          const unsigned char* __restrict__ keep,
                          const float* __restrict__ embed,      // (V,D)
                          unsigned short* __restrict__ x) {     // (B,NA,G,D) bf16
    long long i = (long long)blockIdx.x * blockDim.x + threadIdx.x;
    const long long total = (long long)B_DIM * NA * G_TOK * D_DIM;
    if (i >= total) return;
    int d = (int)(i % D_DIM);
    int g = (int)((i / D_DIM) % G_TOK);
    int n = (int)((i / ((long long)D_DIM * G_TOK)) % NA);
    int b = (int)(i / ((long long)D_DIM * G_TOK * NA));
    int bn = b * NA + n;
    int anc = anchors[bn];
    float val;
    if (g < L_TGT) {                           // target layer g, gather at anchor-1 (clamped)
        int pos = anc - 1; if (pos < 0) pos = 0;
        val = hidden[(((long long)g * B_DIM + b) * S_DIM + pos) * D_DIM + d];
    } else {
        int t = g - L_TGT;
        int tok = MASK_ID;
        if (t == 0 && keep[bn]) {
            int p = anc; if (p < 0) p = 0; if (p > S_DIM - 1) p = S_DIM - 1;
            tok = input_ids[b * S_DIM + p];
        }
        val = embed[(long long)tok * D_DIM + d];
    }
    x[i] = f2bf(val);
}

// ---------------- WMMA bf16 GEMM core ----------------
// A: M x K row-major bf16.  BT: N x K row-major bf16 (B column n is contiguous in k).
// A(16x32) as loaded here (contiguous k, not ISA table order; B matches): lanes 0-15 hold K[kb..kb+15];
// lanes 16-31 hold K[kb+16..kb+31].  B(32x16): lanes 0-15 col n K[kb..kb+15],
// lanes 16-31 same cols K[kb+16..kb+31].  C/D(16x16): vgpr r -> M=r(+8 for hi lanes), N=lane&15.
__device__ __forceinline__ v16bf load_a(const unsigned short* __restrict__ A,
                                        int row, int K, int kb, int sel) {
    const v8bf a0 = *(const v8bf*)(A + (long long)row * K + kb + sel * 16);
    const v8bf a1 = *(const v8bf*)(A + (long long)row * K + kb + sel * 16 + 8);
    v16bf r;
#pragma unroll
    for (int i = 0; i < 8; ++i) { r[i] = a0[i]; r[i + 8] = a1[i]; }
    return r;
}

// Block body: all 256 threads cooperate.  B slab (BN x BK) double-buffered in LDS,
// staged with async global->LDS copies; each wave computes a 32x64 strip (8 WMMAs/k-step).
__device__ __forceinline__ void gemm_block_body(
        const unsigned short* __restrict__ A,
        const unsigned short* __restrict__ BT,
        unsigned short* __restrict__ bsh,      // [2*SLAB] halves
        int K, int Wbase, int Nbase, int lan, int sel, v8f c[2][4]) {
    int t = threadIdx.x;
    int ccol = t >> 2, cchunk = t & 3;         // 64 cols x 4 chunks of 8 halves
    const unsigned short* gsrc = BT + (long long)(Nbase + ccol) * K + cchunk * 8;
    unsigned short* ldst = bsh + (ccol * BK + cchunk * 8);

    copy_slab16(gsrc, ldst);                   // prologue: slab 0 -> buf 0
    const int nsteps = K / BK;
    for (int i = 0; i < nsteps; ++i) {
        int kb = i * BK;
        if (i + 1 < nsteps)
            copy_slab16(gsrc + kb + BK, ldst + ((i + 1) & 1) * SLAB);
        wait_async_le((i + 1 < nsteps) ? 1 : 0);   // slab i landed (newest may be in flight)
        __syncthreads();                            // all waves' copies of slab i visible
        const unsigned short* bs = bsh + (i & 1) * SLAB;
        v16bf a0 = load_a(A, Wbase + lan,      K, kb, sel);
        v16bf a1 = load_a(A, Wbase + 16 + lan, K, kb, sel);
#pragma unroll
        for (int j = 0; j < 4; ++j) {
            v16bf bfr = *(const v16bf*)(bs + (j * 16 + lan) * BK + sel * 16);
            c[0][j] = __builtin_amdgcn_wmma_f32_16x16x32_bf16(
                          false, a0, false, bfr, (short)0, c[0][j], false, false);
            c[1][j] = __builtin_amdgcn_wmma_f32_16x16x32_bf16(
                          false, a1, false, bfr, (short)0, c[1][j], false, false);
        }
        __syncthreads();                            // done reading buf before it is overwritten
    }
}

// GEMM, bf16 output (Q/K/V projections)
__global__ __launch_bounds__(256) void k_gemm_bf16(
        const unsigned short* __restrict__ A, const unsigned short* __restrict__ BT,
        unsigned short* __restrict__ C, int M, int N, int K) {
    __shared__ unsigned short bsh[2 * SLAB];
    int lane = threadIdx.x & 31, wave = threadIdx.x >> 5;
    int bm_cnt = M / BM;
    int bm = blockIdx.x % bm_cnt, ns = blockIdx.x / bm_cnt;
    int Wbase = bm * BM + wave * WM, Nbase = ns * BN;
    int lan = lane & 15, sel = lane >> 4;
    v8f c[2][4] = {};
    gemm_block_body(A, BT, bsh, K, Wbase, Nbase, lan, sel, c);
#pragma unroll
    for (int mi = 0; mi < 2; ++mi)
#pragma unroll
        for (int j = 0; j < 4; ++j)
#pragma unroll
            for (int r = 0; r < 8; ++r) {
                int row = Wbase + mi * 16 + r + sel * 8;
                int col = Nbase + j * 16 + lan;
                C[(long long)row * N + col] = f2bf(c[mi][j][r]);
            }
}

// GEMM for Wo: epilogue fuses q_valid mask and compacts block rows (g>=L)
// into blk (B*NA*BS, D) bf16.  CHS rows (g<L) are dead downstream and dropped.
__global__ __launch_bounds__(256) void k_gemm_wo_blk(
        const unsigned short* __restrict__ A, const unsigned short* __restrict__ BT,
        const unsigned char* __restrict__ keep,
        unsigned short* __restrict__ blk, int M, int N, int K) {
    __shared__ unsigned short bsh[2 * SLAB];
    int lane = threadIdx.x & 31, wave = threadIdx.x >> 5;
    int bm_cnt = M / BM;
    int bm = blockIdx.x % bm_cnt, ns = blockIdx.x / bm_cnt;
    int Wbase = bm * BM + wave * WM, Nbase = ns * BN;
    int lan = lane & 15, sel = lane >> 4;
    v8f c[2][4] = {};
    gemm_block_body(A, BT, bsh, K, Wbase, Nbase, lan, sel, c);
#pragma unroll
    for (int mi = 0; mi < 2; ++mi)
#pragma unroll
        for (int j = 0; j < 4; ++j)
#pragma unroll
            for (int r = 0; r < 8; ++r) {
                int row = Wbase + mi * 16 + r + sel * 8;
                int g  = row % G_TOK;
                int bn = row / G_TOK;
                if (g >= L_TGT) {
                    float val = keep[bn] ? c[mi][j][r] : 0.f;  // q_valid mask (after Wo)
                    int brow = bn * BS + (g - L_TGT);
                    blk[(long long)brow * N + (Nbase + j * 16 + lan)] = f2bf(val);
                }
            }
}

// GEMM, f32 logits straight to d_out (lm_head): A=blk (8192x1024), BT=lm_head bf16 (32000x1024)
__global__ __launch_bounds__(256) void k_gemm_f32(
        const unsigned short* __restrict__ A, const unsigned short* __restrict__ BT,
        float* __restrict__ C, int M, int N, int K) {
    __shared__ unsigned short bsh[2 * SLAB];
    int lane = threadIdx.x & 31, wave = threadIdx.x >> 5;
    int bm_cnt = M / BM;
    int bm = blockIdx.x % bm_cnt, ns = blockIdx.x / bm_cnt;
    int Wbase = bm * BM + wave * WM, Nbase = ns * BN;
    int lan = lane & 15, sel = lane >> 4;
    v8f c[2][4] = {};
    gemm_block_body(A, BT, bsh, K, Wbase, Nbase, lan, sel, c);
#pragma unroll
    for (int mi = 0; mi < 2; ++mi)
#pragma unroll
        for (int j = 0; j < 4; ++j)
#pragma unroll
            for (int r = 0; r < 8; ++r) {
                int row = Wbase + mi * 16 + r + sel * 8;
                int col = Nbase + j * 16 + lan;
                C[(long long)row * N + col] = c[mi][j][r];
            }
}

// ---------------- tiny block-diagonal attention: one block per (group, head) ----------------
__global__ void k_attn(const unsigned short* __restrict__ q,
                       const unsigned short* __restrict__ k,
                       const unsigned short* __restrict__ v,
                       unsigned short* __restrict__ ao) {
    __shared__ float qs[G_TOK][DH], ks[G_TOK][DH], vs[G_TOK][DH];
    int h  = blockIdx.x % NH;
    int bn = blockIdx.x / NH;
    int t  = threadIdx.x;                      // 64 threads == DH
    size_t base = (size_t)bn * G_TOK * D_DIM + h * DH;
    for (int g = 0; g < G_TOK; ++g) {
        qs[g][t] = bf2f(q[base + g * D_DIM + t]);
        ks[g][t] = bf2f(k[base + g * D_DIM + t]);
        vs[g][t] = bf2f(v[base + g * D_DIM + t]);
    }
    __syncthreads();
    if (t < G_TOK) {                           // thread t owns query row g=t
        float sc[G_TOK]; float mx = -3.4e38f;
        for (int m = 0; m < G_TOK; ++m) {
            float s = 0.f;
            for (int d = 0; d < DH; ++d) s += qs[t][d] * ks[m][d];
            s *= 0.125f;                       // 1/sqrt(64)
            sc[m] = s; if (s > mx) mx = s;
        }
        float sum = 0.f;
        for (int m = 0; m < G_TOK; ++m) { float e = __expf(sc[m] - mx); sc[m] = e; sum += e; }
        float inv = 1.f / sum;
        for (int d = 0; d < DH; ++d) {
            float o = 0.f;
            for (int m = 0; m < G_TOK; ++m) o += sc[m] * vs[m][d];
            ao[base + t * D_DIM + d] = f2bf(o * inv);
        }
    }
}

// ---------------- launch ----------------
extern "C" void kernel_launch(void* const* d_in, const int* in_sizes, int n_in,
                              void* d_out, int out_size, void* d_ws, size_t ws_size,
                              hipStream_t stream) {
    const int*   input_ids = (const int*)d_in[0];
    const float* hidden    = (const float*)d_in[1];
    const int*   anchors   = (const int*)d_in[2];
    const unsigned char* keep = (const unsigned char*)d_in[3];
    const float* embed     = (const float*)d_in[4];
    const float* lmw       = (const float*)d_in[5];
    const float* Wq        = (const float*)d_in[6];
    const float* Wk        = (const float*)d_in[7];
    const float* Wv        = (const float*)d_in[8];
    const float* Wo        = (const float*)d_in[9];
    float* logits = (float*)d_out;

    char* ws = (char*)d_ws;
    size_t off = 0;
    auto alloc = [&](size_t bytes) -> unsigned short* {
        unsigned short* p = (unsigned short*)(ws + off);
        off += (bytes + 255) & ~(size_t)255;
        return p;
    };
    const size_t xN   = (size_t)B_DIM * NA * G_TOK * D_DIM;   // 10,485,760
    const size_t blkN = (size_t)B_DIM * NA * BS * D_DIM;      //  8,388,608
    const size_t lmN  = (size_t)V_DIM * D_DIM;                // 32,768,000
    const size_t wN   = (size_t)D_DIM * D_DIM;
    unsigned short* xq   = alloc(xN * 2);
    unsigned short* qb   = alloc(xN * 2);
    unsigned short* kb   = alloc(xN * 2);
    unsigned short* vb   = alloc(xN * 2);
    unsigned short* ao   = alloc(xN * 2);
    unsigned short* blk  = alloc(blkN * 2);
    unsigned short* lmwT = alloc(lmN * 2);
    unsigned short* wqT  = alloc(wN * 2);
    unsigned short* wkT  = alloc(wN * 2);
    unsigned short* wvT  = alloc(wN * 2);
    unsigned short* woT  = alloc(wN * 2);

    // weight conversions (lm_head_w (V,D) is already N-major/K-contiguous for logits=blk@W^T)
    k_cvt  <<<(int)((lmN + 255) / 256), 256, 0, stream>>>(lmw, lmwT, (int)lmN);
    k_cvt_t<<<(int)((wN  + 255) / 256), 256, 0, stream>>>(Wq, wqT, (int)wN);
    k_cvt_t<<<(int)((wN  + 255) / 256), 256, 0, stream>>>(Wk, wkT, (int)wN);
    k_cvt_t<<<(int)((wN  + 255) / 256), 256, 0, stream>>>(Wv, wvT, (int)wN);
    k_cvt_t<<<(int)((wN  + 255) / 256), 256, 0, stream>>>(Wo, woT, (int)wN);

    // build x (gather + noise embed, fused)
    k_build_x<<<(int)((xN + 255) / 256), 256, 0, stream>>>(input_ids, hidden, anchors, keep, embed, xq);

    // Q/K/V projections: M=10240, N=1024, K=1024  (M%256==0, N%64==0)
    const int Mx = B_DIM * NA * G_TOK;
    const int blocks_qkv = (Mx / BM) * (D_DIM / BN);
    k_gemm_bf16<<<blocks_qkv, 256, 0, stream>>>(xq, wqT, qb, Mx, D_DIM, D_DIM);
    k_gemm_bf16<<<blocks_qkv, 256, 0, stream>>>(xq, wkT, kb, Mx, D_DIM, D_DIM);
    k_gemm_bf16<<<blocks_qkv, 256, 0, stream>>>(xq, wvT, vb, Mx, D_DIM, D_DIM);

    // block-diagonal attention: one block per (b,n,head)
    k_attn<<<B_DIM * NA * NH, DH, 0, stream>>>(qb, kb, vb, ao);

    // Wo projection + q_valid mask + compaction to block rows
    k_gemm_wo_blk<<<blocks_qkv, 256, 0, stream>>>(ao, woT, keep, blk, Mx, D_DIM, D_DIM);

    // lm_head logits: M=8192, N=32000, K=1024 -> f32 d_out
    const int Ml = B_DIM * NA * BS;
    const int blocks_l = (Ml / BM) * (V_DIM / BN);
    k_gemm_f32<<<blocks_l, 256, 0, stream>>>(blk, lmwT, logits, Ml, V_DIM, D_DIM);
}
